// prototype_w_7413113553039
// MI455X (gfx1250) — matrix operand order repaired and checked by reading, not compiled, not run on hardware
//
#include <hip/hip_runtime.h>
#include <hip/hip_bf16.h>
#include <math.h>

typedef _Float16 v8h  __attribute__((ext_vector_type(8)));
typedef _Float16 v16h __attribute__((ext_vector_type(16)));
typedef float    v8f  __attribute__((ext_vector_type(8)));

#define NB 16
#define CC 256
#define KK 32
#define LL 4096   // 64*64
#define HP 66     // padded spatial dim (zero halo of 1)
#define PPITCH (HP * HP)   // pixels per image in padded buffer

__device__ __forceinline__ v16h frag_cat(v8h lo, v8h hi) {
    return __builtin_shufflevector(lo, hi, 0,1,2,3,4,5,6,7,8,9,10,11,12,13,14,15);
}

// A fragment (16x32 f16): lane holds row M=lane%16; lanes 0-15: K=[0..7,16..23], lanes 16-31: K=[8..15,24..31]
__device__ __forceinline__ v16h load_A(const _Float16* row, int kb, int lane) {
    int off = ((lane >> 4) & 1) * 8;
    v8h lo = *(const v8h*)(row + kb + off);
    v8h hi = *(const v8h*)(row + kb + off + 16);
    return frag_cat(lo, hi);
}

// B fragment (32x16 f16): lane holds col N=lane%16; lanes 0-15: K=[0..7,16..23], lanes 16-31: K=[8..15,24..31]
__device__ __forceinline__ v16h load_B(const _Float16* col, int kb, int lane) {
    int off = ((lane >> 4) & 1) * 8;
    v8h lo = *(const v8h*)(col + kb + off);
    v8h hi = *(const v8h*)(col + kb + off + 16);
    return frag_cat(lo, hi);
}

__device__ __forceinline__ v8f wmma_f16(v16h a, v16h b, v8f c) {
    return __builtin_amdgcn_wmma_f32_16x16x32_f16(false, a, false, b, (short)0, c, false, false);
}

// ---------------- Stage -1: zero the padded xn buffer (halo must be 0 every call) ----------------
__global__ void zero_kernel(int4* __restrict__ p) {
    p[blockIdx.x * blockDim.x + threadIdx.x] = int4{0, 0, 0, 0};
}

// ---------------- Stage 1: L2-normalize along C; emit f16 copies (padded pixel-major + channel-major) ----------------
__global__ void normalize_kernel(const float* __restrict__ feature,
                                 _Float16* __restrict__ xn_pad,   // (n, 66, 66, c), zero halo
                                 _Float16* __restrict__ xn_ncl,   // (n, c, l)
                                 _Float16* __restrict__ feat_nlc) // (n, l, c)
{
    int idx = blockIdx.x * blockDim.x + threadIdx.x;   // n*L + l, total N*L
    int n = idx >> 12, l = idx & (LL - 1);
    int h = l >> 6, w = l & 63;
    const float* src = feature + (size_t)n * CC * LL + l;
    float ss = 0.f;
    #pragma unroll 4
    for (int c = 0; c < CC; ++c) { float v = src[(size_t)c * LL]; ss += v * v; }
    float inv = 1.0f / fmaxf(sqrtf(ss), 1e-12f);
    _Float16* pl = xn_pad + ((size_t)n * PPITCH + (size_t)(h + 1) * HP + (w + 1)) * CC;
    _Float16* pf = feat_nlc + (size_t)idx * CC;
    #pragma unroll 4
    for (int c = 0; c < CC; ++c) {
        float v = src[(size_t)c * LL];
        float xv = v * inv;
        pl[c] = (_Float16)xv;
        pf[c] = (_Float16)v;
        xn_ncl[((size_t)n * CC + c) * LL + l] = (_Float16)xv;
    }
}

// ---------------- Stage 0: weight conversion / re-layout to f16 ----------------
__global__ void prep_weights(const float* __restrict__ convw, // (K,C,3,3)
                             const float* __restrict__ t1w,   // (C,2C)
                             const float* __restrict__ t2w,   // (C,C)
                             _Float16* __restrict__ convW_h,  // (tap, kout, cin)
                             _Float16* __restrict__ t1A_h,    // (cout, cin)
                             _Float16* __restrict__ t2_h)     // (cout, cin)
{
    int i = blockIdx.x * blockDim.x + threadIdx.x;
    if (i < 9 * KK * CC) {
        int tap = i / (KK * CC); int r = i % (KK * CC); int k = r / CC, c = r % CC;
        int ty = tap / 3, tx = tap % 3;
        convW_h[i] = (_Float16)convw[((k * CC + c) * 3 + ty) * 3 + tx];
    } else {
        int j = i - 9 * KK * CC;
        if (j < CC * CC) {
            int co = j >> 8, ci = j & 255;
            t1A_h[j] = (_Float16)t1w[co * 2 * CC + ci];
        } else {
            int m = j - CC * CC;
            t2_h[m] = (_Float16)t2w[m];
        }
    }
}

// ---------------- Stage 2: 3x3 conv (C->K) as 9 shifted WMMA GEMM taps + softmax over K ----------------
// Padded input => branch-free inner loop: pure b128 loads + 2x wmma per kb step.
__global__ void __launch_bounds__(32)
conv_softmax_kernel(const _Float16* __restrict__ xn_pad,
                    const _Float16* __restrict__ convW_h,
                    const float* __restrict__ conv_b,
                    _Float16* __restrict__ sa_h)  // (n, k, l)
{
    __shared__ float smem[16 * 32];
    int blk = blockIdx.x;                 // n*256 + h*4 + wt
    int n = blk >> 8; int r = blk & 255; int h = r >> 2; int wt = r & 3;
    int lane = threadIdx.x;
    int col = lane & 15;
    int w = wt * 16 + col;
    const _Float16* base = xn_pad + (size_t)n * PPITCH * CC;
    v8f acc0 = {}; v8f acc1 = {};
    #pragma unroll
    for (int tap = 0; tap < 9; ++tap) {
        int dy = tap / 3, dx = tap % 3;   // padded coords: src pixel (h+dy, w+dx)
        const _Float16* bcol  = base + (size_t)((h + dy) * HP + (w + dx)) * CC;
        const _Float16* arow0 = convW_h + (size_t)(tap * KK + col) * CC;
        const _Float16* arow1 = convW_h + (size_t)(tap * KK + 16 + col) * CC;
        for (int kb = 0; kb < CC; kb += 32) {
            __builtin_prefetch(bcol + kb + 64, 0, 1);
            v16h bf = load_B(bcol, kb, lane);
            v16h a0 = load_A(arow0, kb, lane);
            v16h a1 = load_A(arow1, kb, lane);
            acc0 = wmma_f16(a0, bf, acc0);   // out channels 0..15
            acc1 = wmma_f16(a1, bf, acc1);   // out channels 16..31
        }
    }
    // D layout: lane holds col=pixel, rows ch0..ch0+7 per tile half
    int ch0 = (lane < 16) ? 0 : 8;
    #pragma unroll
    for (int i = 0; i < 8; ++i) {
        smem[col * 32 + ch0 + i]      = acc0[i];
        smem[col * 32 + 16 + ch0 + i] = acc1[i];
    }
    __syncthreads();
    if (lane < 16) {
        int l = h * 64 + wt * 16 + lane;
        float vals[32]; float mx = -1e30f;
        #pragma unroll
        for (int k = 0; k < 32; ++k) {
            float v = smem[lane * 32 + k] + conv_b[k];
            vals[k] = v; mx = fmaxf(mx, v);
        }
        float s = 0.f;
        #pragma unroll
        for (int k = 0; k < 32; ++k) { float e = expf(vals[k] - mx); vals[k] = e; s += e; }
        float inv = 1.f / s;
        #pragma unroll
        for (int k = 0; k < 32; ++k)
            sa_h[((size_t)(n * KK + k)) * LL + l] = (_Float16)(vals[k] * inv);
    }
}

// ---------------- Stage 3: rowsum[n,k] = sum_l sa ----------------
__global__ void rowsum_kernel(const _Float16* __restrict__ sa_h, float* __restrict__ rowsum) {
    __shared__ float red[256];
    int nk = blockIdx.x;
    const _Float16* p = sa_h + (size_t)nk * LL;
    float s = 0.f;
    for (int i = threadIdx.x; i < LL; i += 256) s += (float)p[i];
    red[threadIdx.x] = s; __syncthreads();
    for (int st = 128; st > 0; st >>= 1) {
        if (threadIdx.x < st) red[threadIdx.x] += red[threadIdx.x + st];
        __syncthreads();
    }
    if (threadIdx.x == 0) rowsum[nk] = red[0];
}

// ---------------- Stage 4: up^T[n,c,k] = sum_l xn[n,c,l]*sa[n,k,l]  (WMMA, reduce L) ----------------
__global__ void __launch_bounds__(32)
einsum_kernel(const _Float16* __restrict__ xn_ncl,
              const _Float16* __restrict__ sa_h,
              float* __restrict__ up)   // stored (n, k, c)
{
    int blk = blockIdx.x;               // n*32 + mt*2 + nt
    int n = blk >> 5; int r = blk & 31; int mt = r >> 1; int nt = r & 1;
    int lane = threadIdx.x;
    int cRow = mt * 16 + (lane & 15);
    int kCol = nt * 16 + (lane & 15);
    const _Float16* arow = xn_ncl + (size_t)(n * CC + cRow) * LL;
    const _Float16* bcol = sa_h  + (size_t)(n * KK + kCol) * LL;
    v8f acc = {};
    for (int lb = 0; lb < LL; lb += 32) {
        __builtin_prefetch(arow + lb + 128, 0, 1);
        __builtin_prefetch(bcol + lb + 128, 0, 1);
        acc = wmma_f16(load_A(arow, lb, lane), load_B(bcol, lb, lane), acc);
    }
    int c0 = mt * 16 + ((lane < 16) ? 0 : 8);
    float* dst = up + (size_t)(n * KK + kCol) * CC + c0;
    #pragma unroll
    for (int i = 0; i < 8; ++i) dst[i] = acc[i];
}

// ---------------- Stage 5: subtract rowsum*centroid, l2norm over C; rnorm2 for global norm ----------------
__global__ void up_norm1_kernel(const float* __restrict__ centroids,
                                const float* __restrict__ rowsum,
                                float* __restrict__ up,
                                float* __restrict__ rnorm2) {
    __shared__ float red[256];
    int nk = blockIdx.x; int k = nk & 31;
    int c = threadIdx.x;
    float rs = rowsum[nk];
    float v = up[(size_t)nk * CC + c] - rs * centroids[k * CC + c];
    red[c] = v * v; __syncthreads();
    for (int st = 128; st > 0; st >>= 1) {
        if (c < st) red[c] += red[c + st];
        __syncthreads();
    }
    float norm = sqrtf(red[0]);
    float inv = 1.f / fmaxf(norm, 1e-12f);
    up[(size_t)nk * CC + c] = v * inv;
    if (c == 0) { float t = norm * inv; rnorm2[nk] = t * t; }
}

// ---------------- Stage 6: global l2norm over K*C + fc (C x 8192) ----------------
__global__ void upfc_kernel(const float* __restrict__ upfc_w,
                            const float* __restrict__ upfc_b,
                            const float* __restrict__ up,
                            const float* __restrict__ rnorm2,
                            float* __restrict__ upvec) {
    __shared__ float uflat[KK * CC];
    __shared__ float gsh;
    int n = blockIdx.x; int t = threadIdx.x;
    for (int i = t; i < KK * CC; i += 256) uflat[i] = up[(size_t)n * KK * CC + i];
    if (t == 0) {
        float s = 0.f;
        for (int k = 0; k < KK; ++k) s += rnorm2[n * KK + k];
        gsh = 1.f / fmaxf(sqrtf(s), 1e-12f);
    }
    __syncthreads();
    float g = gsh;
    const float* wrow = upfc_w + (size_t)t * (KK * CC);
    float acc = 0.f;
    #pragma unroll 4
    for (int j = 0; j < KK * CC; ++j) acc += wrow[j] * uflat[j];
    upvec[n * CC + t] = acc * g + upfc_b[t];
}

// ---------------- Stage 7: bias1[n,c] = t1_b[c] + t1_w[c, C:2C] . upvec[n] ----------------
__global__ void bias1_kernel(const float* __restrict__ t1w,
                             const float* __restrict__ t1b,
                             const float* __restrict__ upvec,
                             float* __restrict__ bias1) {
    __shared__ float uv[CC];
    int n = blockIdx.x; int c = threadIdx.x;
    uv[c] = upvec[n * CC + c]; __syncthreads();
    const float* wrow = t1w + (size_t)c * (2 * CC) + CC;
    float acc = t1b[c];
    #pragma unroll 4
    for (int j = 0; j < CC; ++j) acc += wrow[j] * uv[j];
    bias1[n * CC + c] = acc;
}

// ---------------- Stage 8: t1 GEMM: h = relu(t1A @ feature + bias1), f16 out, 4 N-tiles/wave ----------------
__global__ void __launch_bounds__(32)
t1_kernel(const _Float16* __restrict__ t1A_h,
          const _Float16* __restrict__ feat_nlc,
          const float* __restrict__ bias1,
          _Float16* __restrict__ h_nlc) {
    int blk = blockIdx.x;               // n*(16*64) + mt*64 + lt4
    int lt4 = blk & 63; int r = blk >> 6; int mt = r & 15; int n = r >> 4;
    int lane = threadIdx.x;
    int row = mt * 16 + (lane & 15);
    int pix0 = lt4 * 64 + (lane & 15);
    const _Float16* arow = t1A_h + (size_t)row * CC;
    v8f acc[4] = {{}, {}, {}, {}};
    for (int kb = 0; kb < CC; kb += 32) {
        v16h a = load_A(arow, kb, lane);
        #pragma unroll
        for (int j = 0; j < 4; ++j) {
            const _Float16* bcol = feat_nlc + (size_t)(n * LL + pix0 + j * 16) * CC;
            acc[j] = wmma_f16(a, load_B(bcol, kb, lane), acc[j]);
        }
    }
    int c0 = mt * 16 + ((lane < 16) ? 0 : 8);
    const float* bp = bias1 + n * CC + c0;
    float b[8];
    #pragma unroll
    for (int i = 0; i < 8; ++i) b[i] = bp[i];
    #pragma unroll
    for (int j = 0; j < 4; ++j) {
        v8h out;
        #pragma unroll
        for (int i = 0; i < 8; ++i) out[i] = (_Float16)fmaxf(acc[j][i] + b[i], 0.f);
        *(v8h*)(h_nlc + (size_t)(n * LL + pix0 + j * 16) * CC + c0) = out;
    }
}

// ---------------- Stage 9: t2 GEMM: out = relu(t2 @ h + t2_b), f32 NCHW out ----------------
__global__ void __launch_bounds__(32)
t2_kernel(const _Float16* __restrict__ t2_h,
          const _Float16* __restrict__ h_nlc,
          const float* __restrict__ t2b,
          float* __restrict__ out) {
    int blk = blockIdx.x;
    int lt4 = blk & 63; int r = blk >> 6; int mt = r & 15; int n = r >> 4;
    int lane = threadIdx.x;
    int row = mt * 16 + (lane & 15);
    int pix0 = lt4 * 64 + (lane & 15);
    const _Float16* arow = t2_h + (size_t)row * CC;
    v8f acc[4] = {{}, {}, {}, {}};
    for (int kb = 0; kb < CC; kb += 32) {
        v16h a = load_A(arow, kb, lane);
        #pragma unroll
        for (int j = 0; j < 4; ++j) {
            const _Float16* bcol = h_nlc + (size_t)(n * LL + pix0 + j * 16) * CC;
            acc[j] = wmma_f16(a, load_B(bcol, kb, lane), acc[j]);
        }
    }
    int c0 = mt * 16 + ((lane < 16) ? 0 : 8);
    float b[8];
    #pragma unroll
    for (int i = 0; i < 8; ++i) b[i] = t2b[c0 + i];
    #pragma unroll
    for (int j = 0; j < 4; ++j) {
        int pix = pix0 + j * 16;
        #pragma unroll
        for (int i = 0; i < 8; ++i) {
            float v = fmaxf(acc[j][i] + b[i], 0.f);
            out[((size_t)(n * CC + c0 + i)) * LL + pix] = v;
        }
    }
}

extern "C" void kernel_launch(void* const* d_in, const int* in_sizes, int n_in,
                              void* d_out, int out_size, void* d_ws, size_t ws_size,
                              hipStream_t stream) {
    const float* feature   = (const float*)d_in[0];
    const float* conv_up_w = (const float*)d_in[1];
    const float* conv_up_b = (const float*)d_in[2];
    const float* centroids = (const float*)d_in[3];
    const float* upfc_w    = (const float*)d_in[4];
    const float* upfc_b    = (const float*)d_in[5];
    const float* t1_w      = (const float*)d_in[6];
    const float* t1_b      = (const float*)d_in[7];
    const float* t2_w      = (const float*)d_in[8];
    const float* t2_b      = (const float*)d_in[9];
    float* out = (float*)d_out;

    uint8_t* ws = (uint8_t*)d_ws;
    size_t off = 0;
    auto alloc = [&](size_t bytes) -> void* {
        void* p = ws + off;
        off = (off + bytes + 255) & ~(size_t)255;
        return p;
    };
    const size_t xn_pad_bytes = (size_t)NB * PPITCH * CC * 2;   // 35,684,352 B
    _Float16* xn_pad   = (_Float16*)alloc(xn_pad_bytes);
    _Float16* xn_ncl   = (_Float16*)alloc((size_t)NB * CC * LL * 2);
    _Float16* feat_nlc = (_Float16*)alloc((size_t)NB * LL * CC * 2);
    _Float16* h_nlc    = (_Float16*)alloc((size_t)NB * LL * CC * 2);
    _Float16* sa_h     = (_Float16*)alloc((size_t)NB * KK * LL * 2);
    _Float16* convW_h  = (_Float16*)alloc((size_t)9 * KK * CC * 2);
    _Float16* t1A_h    = (_Float16*)alloc((size_t)CC * CC * 2);
    _Float16* t2w_h    = (_Float16*)alloc((size_t)CC * CC * 2);
    float*    rowsum   = (float*)alloc((size_t)NB * KK * 4);
    float*    rnorm2   = (float*)alloc((size_t)NB * KK * 4);
    float*    up       = (float*)alloc((size_t)NB * KK * CC * 4);
    float*    upvec    = (float*)alloc((size_t)NB * CC * 4);
    float*    bias1    = (float*)alloc((size_t)NB * CC * 4);

    // Zero padded buffer (halo must be zero; interior overwritten by normalize).
    const int n4 = (int)(xn_pad_bytes / 16);                    // 2,230,272 int4s
    zero_kernel<<<dim3(n4 / 256), dim3(256), 0, stream>>>((int4*)xn_pad);

    prep_weights<<<dim3((9 * KK * CC + 2 * CC * CC) / 256), dim3(256), 0, stream>>>(
        conv_up_w, t1_w, t2_w, convW_h, t1A_h, t2w_h);
    normalize_kernel<<<dim3(NB * LL / 256), dim3(256), 0, stream>>>(
        feature, xn_pad, xn_ncl, feat_nlc);
    conv_softmax_kernel<<<dim3(NB * 64 * 4), dim3(32), 0, stream>>>(
        xn_pad, convW_h, conv_up_b, sa_h);
    rowsum_kernel<<<dim3(NB * KK), dim3(256), 0, stream>>>(sa_h, rowsum);
    einsum_kernel<<<dim3(NB * 32), dim3(32), 0, stream>>>(xn_ncl, sa_h, up);
    up_norm1_kernel<<<dim3(NB * KK), dim3(256), 0, stream>>>(centroids, rowsum, up, rnorm2);
    upfc_kernel<<<dim3(NB), dim3(256), 0, stream>>>(upfc_w, upfc_b, up, rnorm2, upvec);
    bias1_kernel<<<dim3(NB), dim3(256), 0, stream>>>(t1_w, t1_b, upvec, bias1);
    t1_kernel<<<dim3(NB * 16 * 64), dim3(32), 0, stream>>>(t1A_h, feat_nlc, bias1, h_nlc);
    t2_kernel<<<dim3(NB * 16 * 64), dim3(32), 0, stream>>>(t2w_h, h_nlc, t2_b, out);

    (void)in_sizes; (void)n_in; (void)out_size; (void)ws_size;
}
